// RefactoredMHA_79740362817870
// MI455X (gfx1250) — matrix operand order repaired and checked by reading, not compiled, not run on hardware
//
#include <hip/hip_runtime.h>
#include <math.h>

typedef __attribute__((ext_vector_type(16))) __bf16 v16bf;
typedef __attribute__((ext_vector_type(8)))  __bf16 v8bf;
typedef __attribute__((ext_vector_type(8)))  float  v8f;
typedef __attribute__((ext_vector_type(4)))  float  f32x4;
typedef __attribute__((ext_vector_type(4)))  unsigned int u32x4;
typedef __attribute__((ext_vector_type(4)))  int      intx4;
typedef unsigned short ushort16_t;

#define BATCH 4
#define SEQ   4096
#define EMB   512
#define TILE  128
#define KSTEP 32
#define LDT   40            // padded LDS row stride (elements); 80B rows, 16B aligned
#define BUFSZ (TILE * LDT)  // one LDS buffer (elements)

#if __has_builtin(__builtin_amdgcn_global_load_async_to_lds_b128)
#define HAVE_ASYNC_LDS 1
#else
#define HAVE_ASYNC_LDS 0
#endif

#define GPTR(p) ((__attribute__((address_space(1))) intx4*)(p))
#define LPTR(p) ((__attribute__((address_space(3))) intx4*)(p))

__device__ __forceinline__ void wait_async0() {
#if __has_builtin(__builtin_amdgcn_s_wait_asynccnt)
  __builtin_amdgcn_s_wait_asynccnt(0);
#else
  asm volatile("s_wait_asynccnt 0x0" ::: "memory");
#endif
}

// ---------- fp32 -> bf16 round-to-nearest-even ----------
__device__ __forceinline__ ushort16_t f2bf(float f) {
  union { float f; unsigned u; } v; v.f = f;
  unsigned u = v.u;
  unsigned r = u + 0x7FFFu + ((u >> 16) & 1u);
  return (ushort16_t)(r >> 16);
}

// ---------- staging: 256 threads handle a 128x32 tile; 16 elements/thread ----------
__device__ __forceinline__ void gload_bf16(const ushort16_t* __restrict__ src, int ld,
                                           int row0, int k0, int tid,
                                           u32x4& r0, u32x4& r1) {
  int row = tid >> 1, col = (tid & 1) << 4;
  const ushort16_t* s = src + (long)(row0 + row) * ld + k0 + col;
  r0 = *(const u32x4*)s;
  r1 = *(const u32x4*)(s + 8);
}

__device__ __forceinline__ void lstore_bf16(ushort16_t* lds, int tid, u32x4 r0, u32x4 r1) {
  int row = tid >> 1, col = (tid & 1) << 4;
  ushort16_t* d = lds + row * LDT + col;
  *(u32x4*)d       = r0;
  *(u32x4*)(d + 8) = r1;
}

#if HAVE_ASYNC_LDS
// async DMA: global -> LDS, 16B per lane per op, tracked on ASYNCcnt
__device__ __forceinline__ void async_stage(const ushort16_t* __restrict__ src, int ld,
                                            int row0, int k0, int tid,
                                            ushort16_t* ldsbuf) {
  int row = tid >> 1, col = (tid & 1) << 4;
  const ushort16_t* s = src + (long)(row0 + row) * ld + k0 + col;
  ushort16_t* d = ldsbuf + row * LDT + col;
  __builtin_amdgcn_global_load_async_to_lds_b128(GPTR(s),     LPTR(d),     0, 0);
  __builtin_amdgcn_global_load_async_to_lds_b128(GPTR(s + 8), LPTR(d + 8), 0, 0);
}
#endif

__device__ __forceinline__ void gload_f32(const float* __restrict__ src, int ld,
                                          int row0, int k0, int tid, f32x4 r[4]) {
  int row = tid >> 1, col = (tid & 1) << 4;
  const float* s = src + (long)(row0 + row) * ld + k0 + col;
#pragma unroll
  for (int j = 0; j < 4; ++j) r[j] = *(const f32x4*)(s + 4 * j);
}

__device__ __forceinline__ void lstore_f32(ushort16_t* lds, int tid, const f32x4 r[4]) {
  int row = tid >> 1, col = (tid & 1) << 4;
  ushort16_t* d = lds + row * LDT + col;
  union { ushort16_t us[16]; u32x4 v[2]; } pk;
#pragma unroll
  for (int j = 0; j < 4; ++j)
#pragma unroll
    for (int i = 0; i < 4; ++i) pk.us[j * 4 + i] = f2bf(r[j][i]);
  *(u32x4*)d       = pk.v[0];
  *(u32x4*)(d + 8) = pk.v[1];
}

// ---------- one K-step: 8 waves, each 16(M)x128(N); preload B frags, then 8 WMMAs ----
// A frag (16x32): lane g=lane>>4, m=lane&15; elems 0..7 -> K=8g+i, 8..15 -> K=16+8g+(i-8)
// B frag (32x16): lane h=lane>>4, n=lane&15; elems 0..7 -> K=8h+i, 8..15 -> K=16+8h+(i-8)
__device__ __forceinline__ void wave_compute(const ushort16_t* Alds,
                                             const ushort16_t* Blds,
                                             int tid, v8f acc[8]) {
  int wv = tid >> 5, lane = tid & 31;
  int g = lane >> 4, ln = lane & 15;
  const ushort16_t* Ar = Alds + (wv * 16 + ln) * LDT + 8 * g;
  v8bf aLo = *(const v8bf*)Ar;
  v8bf aHi = *(const v8bf*)(Ar + 16);
  v16bf a = __builtin_shufflevector(aLo, aHi, 0,1,2,3,4,5,6,7,8,9,10,11,12,13,14,15);
  v16bf bf[8];
#pragma unroll
  for (int nt = 0; nt < 8; ++nt) {
    const ushort16_t* Br = Blds + (nt * 16 + ln) * LDT + 8 * g;
    v8bf bLo = *(const v8bf*)Br;
    v8bf bHi = *(const v8bf*)(Br + 16);
    bf[nt] = __builtin_shufflevector(bLo, bHi, 0,1,2,3,4,5,6,7,8,9,10,11,12,13,14,15);
  }
#pragma unroll
  for (int nt = 0; nt < 8; ++nt)
    acc[nt] = __builtin_amdgcn_wmma_f32_16x16x32_bf16(
        false, a, false, bf[nt], (short)0, acc[nt], false, false);
}

// ---------- double-buffered GEMM main loop (bf16 sources) ----------
__device__ __forceinline__ void gemm_loop_bf16(
    const ushort16_t* __restrict__ A, int lda, int row0,
    const ushort16_t* __restrict__ Bm, int ldb, int col0,
    int kd, int tid, ushort16_t* Alds, ushort16_t* Blds, v8f acc[8]) {
#if HAVE_ASYNC_LDS
  // Prologue: DMA tile 0 into buf0.
  async_stage(A, lda, row0, 0, tid, Alds);
  async_stage(Bm, ldb, col0, 0, tid, Blds);
  wait_async0();
  __syncthreads();
  int cur = 0;
  for (int k0 = 0; k0 < kd; k0 += KSTEP) {
    ushort16_t* Ab = Alds + cur * BUFSZ;
    ushort16_t* Bb = Blds + cur * BUFSZ;
    if (k0 + KSTEP < kd) {  // DMA next tile while this one is consumed
      async_stage(A, lda, row0, k0 + KSTEP, tid, Alds + (cur ^ 1) * BUFSZ);
      async_stage(Bm, ldb, col0, k0 + KSTEP, tid, Blds + (cur ^ 1) * BUFSZ);
    }
    wave_compute(Ab, Bb, tid, acc);
    wait_async0();     // this wave's DMA done
    __syncthreads();   // all waves' DMA + reads done
    cur ^= 1;
  }
#else
  u32x4 a0, a1, b0, b1;
  gload_bf16(A, lda, row0, 0, tid, a0, a1);
  gload_bf16(Bm, ldb, col0, 0, tid, b0, b1);
  int cur = 0;
  for (int k0 = 0; k0 < kd; k0 += KSTEP) {
    ushort16_t* Ab = Alds + cur * BUFSZ;
    ushort16_t* Bb = Blds + cur * BUFSZ;
    lstore_bf16(Ab, tid, a0, a1);
    lstore_bf16(Bb, tid, b0, b1);
    __syncthreads();
    if (k0 + KSTEP < kd) {
      gload_bf16(A, lda, row0, k0 + KSTEP, tid, a0, a1);
      gload_bf16(Bm, ldb, col0, k0 + KSTEP, tid, b0, b1);
    }
    wave_compute(Ab, Bb, tid, acc);
    cur ^= 1;
  }
#endif
}

// ---------- kernel 1: projection  Y = x @ W^T + b  (emit bf16; V transposed) ----------
__global__ __launch_bounds__(256) void proj_kernel(
    const float* __restrict__ x,    // [B*S, E] fp32
    const float* __restrict__ W,    // [E, E] fp32 (row = out feature)
    const float* __restrict__ bias, // [E]
    ushort16_t* __restrict__ outRM, // bf16 [B*S, E]  (q or k)  or null
    ushort16_t* __restrict__ outVT) // bf16 [B][E][S] (v transposed) or null
{
  __shared__ ushort16_t Alds[2 * BUFSZ];
  __shared__ ushort16_t Blds[2 * BUFSZ];
  int tid = threadIdx.x;
  int row0 = blockIdx.x * TILE;   // token rows
  int col0 = blockIdx.y * TILE;   // output features
  v8f zero = {0.f,0.f,0.f,0.f,0.f,0.f,0.f,0.f};
  v8f acc[8];
#pragma unroll
  for (int i = 0; i < 8; ++i) acc[i] = zero;

  f32x4 ra[4], rb[4];
  gload_f32(x, EMB, row0, 0, tid, ra);
  gload_f32(W, EMB, col0, 0, tid, rb);
  int cur = 0;
  for (int k0 = 0; k0 < EMB; k0 += KSTEP) {
    ushort16_t* Ab = Alds + cur * BUFSZ;
    ushort16_t* Bb = Blds + cur * BUFSZ;
    lstore_f32(Ab, tid, ra);
    lstore_f32(Bb, tid, rb);
    __syncthreads();
    if (k0 + KSTEP < EMB) {
      gload_f32(x, EMB, row0, k0 + KSTEP, tid, ra);
      gload_f32(W, EMB, col0, k0 + KSTEP, tid, rb);
    }
    wave_compute(Ab, Bb, tid, acc);
    cur ^= 1;
  }

  int wv = tid >> 5, lane = tid & 31, h = lane >> 4, ln = lane & 15;
  int m0 = row0 + wv * 16 + 8 * h;   // 8 consecutive rows r=0..7
  if (outRM) {
#pragma unroll
    for (int nt = 0; nt < 8; ++nt) {
      int n = col0 + nt * 16 + ln;
      float bv = bias[n];
#pragma unroll
      for (int r = 0; r < 8; ++r)
        outRM[(long)(m0 + r) * EMB + n] = f2bf(acc[nt][r] + bv);
    }
  } else {
    int b = m0 / SEQ, s0 = m0 % SEQ;  // 8 rows stay within one batch (m0 % 8 == 0)
#pragma unroll
    for (int nt = 0; nt < 8; ++nt) {
      int n = col0 + nt * 16 + ln;
      float bv = bias[n];
      union { ushort16_t us[8]; u32x4 v4; } pk;
#pragma unroll
      for (int r = 0; r < 8; ++r) pk.us[r] = f2bf(acc[nt][r] + bv);
      *(u32x4*)&outVT[((long)b * EMB + n) * SEQ + s0] = pk.v4;  // 16B store
    }
  }
}

// ---------- kernel 2: scores = (q @ k^T) * scale ----------
__global__ __launch_bounds__(256) void scores_kernel(
    const ushort16_t* __restrict__ q,  // bf16 [S, E] (one batch)
    const ushort16_t* __restrict__ k,  // bf16 [S, E]
    float* __restrict__ sc,            // f32 [S, S]
    float scale)
{
  __shared__ ushort16_t Alds[2 * BUFSZ];
  __shared__ ushort16_t Blds[2 * BUFSZ];
  int tid = threadIdx.x;
  int row0 = blockIdx.x * TILE;   // query rows
  int col0 = blockIdx.y * TILE;   // key rows
  v8f zero = {0.f,0.f,0.f,0.f,0.f,0.f,0.f,0.f};
  v8f acc[8];
#pragma unroll
  for (int i = 0; i < 8; ++i) acc[i] = zero;

  gemm_loop_bf16(q, EMB, row0, k, EMB, col0, EMB, tid, Alds, Blds, acc);

  int wv = tid >> 5, lane = tid & 31, h = lane >> 4, ln = lane & 15;
  int m0 = row0 + wv * 16 + 8 * h;
#pragma unroll
  for (int nt = 0; nt < 8; ++nt) {
    int n = col0 + nt * 16 + ln;
#pragma unroll
    for (int r = 0; r < 8; ++r)
      sc[(long)(m0 + r) * SEQ + n] = acc[nt][r] * scale;
  }
}

// ---------- kernel 3: row softmax, f32 scores -> bf16 probs ----------
__global__ __launch_bounds__(256) void softmax_kernel(
    const float* __restrict__ sc, ushort16_t* __restrict__ pr)
{
  __shared__ float red[256];
  int tid = threadIdx.x;
  int row = blockIdx.x;
  const float* srow = sc + (long)row * SEQ;
  float mx = -INFINITY;
  for (int j = tid; j < SEQ; j += 256) mx = fmaxf(mx, srow[j]);
  red[tid] = mx; __syncthreads();
  for (int s = 128; s > 0; s >>= 1) {
    if (tid < s) red[tid] = fmaxf(red[tid], red[tid + s]);
    __syncthreads();
  }
  mx = red[0]; __syncthreads();
  float sum = 0.f;
  for (int j = tid; j < SEQ; j += 256) sum += __expf(srow[j] - mx);
  red[tid] = sum; __syncthreads();
  for (int s = 128; s > 0; s >>= 1) {
    if (tid < s) red[tid] += red[tid + s];
    __syncthreads();
  }
  float inv = 1.f / red[0];
  ushort16_t* prow = pr + (long)row * SEQ;
  for (int j = tid; j < SEQ; j += 256)
    prow[j] = f2bf(__expf(srow[j] - mx) * inv);
}

// ---------- kernel 4: out = probs @ V  (V pre-transposed [E][S]) ----------
__global__ __launch_bounds__(256) void pv_kernel(
    const ushort16_t* __restrict__ pr,  // bf16 [S, S]
    const ushort16_t* __restrict__ vT,  // bf16 [E, S]
    float* __restrict__ out)            // f32 [S, E]
{
  __shared__ ushort16_t Alds[2 * BUFSZ];
  __shared__ ushort16_t Blds[2 * BUFSZ];
  int tid = threadIdx.x;
  int row0 = blockIdx.x * TILE;   // query rows
  int col0 = blockIdx.y * TILE;   // embed cols
  v8f zero = {0.f,0.f,0.f,0.f,0.f,0.f,0.f,0.f};
  v8f acc[8];
#pragma unroll
  for (int i = 0; i < 8; ++i) acc[i] = zero;

  gemm_loop_bf16(pr, SEQ, row0, vT, SEQ, col0, SEQ, tid, Alds, Blds, acc);

  int wv = tid >> 5, lane = tid & 31, h = lane >> 4, ln = lane & 15;
  int m0 = row0 + wv * 16 + 8 * h;
#pragma unroll
  for (int nt = 0; nt < 8; ++nt) {
    int n = col0 + nt * 16 + ln;
#pragma unroll
    for (int r = 0; r < 8; ++r)
      out[(long)(m0 + r) * EMB + n] = acc[nt][r];
  }
}

extern "C" void kernel_launch(void* const* d_in, const int* in_sizes, int n_in,
                              void* d_out, int out_size, void* d_ws, size_t ws_size,
                              hipStream_t stream) {
  (void)in_sizes; (void)n_in; (void)out_size; (void)ws_size;
  const float* x  = (const float*)d_in[0];
  // d_in[1] = step (==0), unused
  const float* Wq = (const float*)d_in[2];
  const float* bq = (const float*)d_in[3];
  const float* Wk = (const float*)d_in[4];
  const float* bk = (const float*)d_in[5];
  const float* Wv = (const float*)d_in[6];
  const float* bv = (const float*)d_in[7];
  float* out = (float*)d_out;

  // workspace layout (all 16B-aligned):
  //   qb  bf16 [B*S, E]          16 MB
  //   kb  bf16 [B*S, E]          16 MB
  //   vT  bf16 [B][E][S]         16 MB
  //   sc  f32  [S, S]            64 MB  (reused per batch)
  //   pr  bf16 [S, S]            32 MB  (reused per batch)
  ushort16_t* qb = (ushort16_t*)d_ws;
  ushort16_t* kb = qb + (long)BATCH * SEQ * EMB;
  ushort16_t* vT = kb + (long)BATCH * SEQ * EMB;
  float*      sc = (float*)(vT + (long)BATCH * EMB * SEQ);
  ushort16_t* pr = (ushort16_t*)(sc + (long)SEQ * SEQ);

  dim3 blk(256);
  dim3 gproj(BATCH * SEQ / TILE, EMB / TILE);           // 128 x 4
  proj_kernel<<<gproj, blk, 0, stream>>>(x, Wq, bq, qb, nullptr);
  proj_kernel<<<gproj, blk, 0, stream>>>(x, Wk, bk, kb, nullptr);
  proj_kernel<<<gproj, blk, 0, stream>>>(x, Wv, bv, nullptr, vT);

  const float scale = 0.04419417382415922f;  // 1 / sqrt(512)
  for (int b = 0; b < BATCH; ++b) {
    scores_kernel<<<dim3(SEQ / TILE, SEQ / TILE), blk, 0, stream>>>(
        qb + (long)b * SEQ * EMB, kb + (long)b * SEQ * EMB, sc, scale);
    softmax_kernel<<<dim3(SEQ), blk, 0, stream>>>(sc, pr);
    pv_kernel<<<dim3(SEQ / TILE, EMB / TILE), blk, 0, stream>>>(
        pr, vT + (long)b * EMB * SEQ, out + (long)b * SEQ * EMB);
  }
}
